// ColumnarNetwork_30408368455888
// MI455X (gfx1250) — 8-bit matrix operand order repaired and checked by reading, compiled, not run on hardware
//
#include <hip/hip_runtime.h>
#include <hip/hip_bf16.h>

typedef __attribute__((ext_vector_type(8))) int v8i;

namespace {
constexpr int NB  = 16;
constexpr int C   = 128;
constexpr int T   = 8;
constexpr int BR  = 16;
constexpr int S   = 8;
constexpr int SY  = 64;
constexpr int NIN = 16384;
constexpr int TPB_BR    = 4;
constexpr int SEG_PER_T = S / TPB_BR;
constexpr int BR_PER_BLK = 256 / TPB_BR;
}

__global__ __launch_bounds__(256) void k_bitmask(const float* __restrict__ x,
                                                 unsigned short* __restrict__ bm) {
    int n = blockIdx.x * 256 + threadIdx.x;
    unsigned m = 0;
#pragma unroll
    for (int b = 0; b < NB; ++b)
        m |= (x[b * NIN + n] != 0.0f ? 1u : 0u) << b;
    bm[n] = (unsigned short)m;
}

__device__ __forceinline__ unsigned spread4(unsigned nib) {
    return __umul24(nib, 0x00204081u) & 0x01010101u;
}
__device__ __forceinline__ unsigned pack4(unsigned bytes01) {
    return (bytes01 * 0x01020408u) >> 24 & 0xFu;
}

template <int LAYER>
__global__ __launch_bounds__(256) void k_layer(const unsigned short* __restrict__ bm_g,
                                               const int* __restrict__ idx,
                                               unsigned short* __restrict__ am_out,
                                               int* __restrict__ out_i32) {
    __shared__ __align__(16) unsigned short lds_bm[NIN];
    __shared__ __align__(16) unsigned short lds_stage[BR_PER_BLK * S];

    const int tid = threadIdx.x;

    {
        const uint4* src = (const uint4*)bm_g;
        uint4* dst = (uint4*)lds_bm;
#pragma unroll
        for (int j = 0; j < (NIN * 2 / 16) / 256; ++j)
            dst[j * 256 + tid] = src[j * 256 + tid];
    }
    __syncthreads();

    const int branchLocal = tid / TPB_BR;
    const int sub = tid % TPB_BR;
    const int branchG = blockIdx.x * BR_PER_BLK + branchLocal;
    int base;
    if (LAYER == 1) {
        base = branchG * (S * SY);
    } else {
        int c = branchG >> 4, br = branchG & 15;
        base = (c * (T * BR) + br) * (S * SY);
    }
    const int4* ip = (const int4*)(idx + base) + sub * SEG_PER_T * (SY / 4);

#pragma unroll
    for (int ss = 0; ss < SEG_PER_T; ++ss) {
        unsigned a0 = 0, a1 = 0, a2 = 0, a3 = 0;
#pragma unroll
        for (int g = 0; g < SY / 4; ++g) {
            int4 iv = ip[ss * (SY / 4) + g];
            int i0 = iv.x, i1 = iv.y, i2 = iv.z, i3 = iv.w;
            unsigned m0 = lds_bm[i0 < 0 ? 0 : i0]; if (i0 < 0) m0 = 0;
            unsigned m1 = lds_bm[i1 < 0 ? 0 : i1]; if (i1 < 0) m1 = 0;
            unsigned m2 = lds_bm[i2 < 0 ? 0 : i2]; if (i2 < 0) m2 = 0;
            unsigned m3 = lds_bm[i3 < 0 ? 0 : i3]; if (i3 < 0) m3 = 0;
            a0 += spread4(m0 & 0xFu);  a1 += spread4((m0 >> 4) & 0xFu);
            a2 += spread4((m0 >> 8) & 0xFu);  a3 += spread4(m0 >> 12);
            a0 += spread4(m1 & 0xFu);  a1 += spread4((m1 >> 4) & 0xFu);
            a2 += spread4((m1 >> 8) & 0xFu);  a3 += spread4(m1 >> 12);
            a0 += spread4(m2 & 0xFu);  a1 += spread4((m2 >> 4) & 0xFu);
            a2 += spread4((m2 >> 8) & 0xFu);  a3 += spread4(m2 >> 12);
            a0 += spread4(m3 & 0xFu);  a1 += spread4((m3 >> 4) & 0xFu);
            a2 += spread4((m3 >> 8) & 0xFu);  a3 += spread4(m3 >> 12);
        }
        unsigned t0 = ((a0 + 0x70707070u) >> 7) & 0x01010101u;
        unsigned t1 = ((a1 + 0x70707070u) >> 7) & 0x01010101u;
        unsigned t2 = ((a2 + 0x70707070u) >> 7) & 0x01010101u;
        unsigned t3 = ((a3 + 0x70707070u) >> 7) & 0x01010101u;
        unsigned mask = pack4(t0) | (pack4(t1) << 4) | (pack4(t2) << 8) | (pack4(t3) << 12);
        lds_stage[branchLocal * S + sub * SEG_PER_T + ss] = (unsigned short)mask;
    }
    __syncthreads();

    const int lane = tid & 31;
    const int wave = tid >> 5;
    const int M = lane & 15;

    v8i bsel;
#pragma unroll
    for (int v = 0; v < 8; ++v) {
        int kb = (v >> 1) * 16 + ((lane >= 16) ? 8 : 0) + (v & 1) * 4;
        unsigned w = 0;
#pragma unroll
        for (int j = 0; j < 4; ++j)
            if (((kb + j) >> 3) == M) w |= 1u << (8 * j);
        bsel[v] = (int)w;
    }

    const unsigned short* mstage = &lds_stage[wave * 8 * S];
    v8i amat;
#pragma unroll
    for (int r = 0; r < 4; ++r) {
        int kstart = r * 16 + ((lane >= 16) ? 8 : 0);
        uint4 md = *(const uint4*)(mstage + kstart);
        unsigned t, p, q0, q1, q2, q3;
        t = md.x >> M; p = t & 0x00010001u; q0 = (p | (p >> 8)) & 0x0101u;
        t = md.y >> M; p = t & 0x00010001u; q1 = (p | (p >> 8)) & 0x0101u;
        t = md.z >> M; p = t & 0x00010001u; q2 = (p | (p >> 8)) & 0x0101u;
        t = md.w >> M; p = t & 0x00010001u; q3 = (p | (p >> 8)) & 0x0101u;
        amat[2 * r]     = (int)(q0 | (q1 << 16));
        amat[2 * r + 1] = (int)(q2 | (q3 << 16));
    }

    v8i cz = {0, 0, 0, 0, 0, 0, 0, 0};
    v8i d = __builtin_amdgcn_wmma_i32_16x16x64_iu8(false, amat, false, bsel, cz,
                                                   false, false);

    if (LAYER == 1) {
        unsigned bits = 0;
#pragma unroll
        for (int i = 0; i < 8; ++i) bits |= (d[i] >= 4 ? 1u : 0u) << i;
        unsigned other = __shfl(bits, (lane + 16) & 31, 32);
        if (lane < 8)
            am_out[blockIdx.x * BR_PER_BLK + wave * 8 + lane] =
                (unsigned short)(bits | (other << 8));
    } else {
        if (M < 8) {
            int brG = blockIdx.x * BR_PER_BLK + wave * 8 + M;
            int c = brG >> 4, br = brG & 15;
            int bOff = (lane < 16) ? 0 : 8;
#pragma unroll
            for (int i = 0; i < 8; ++i)
                out_i32[((bOff + i) * C + c) * BR + br] = (d[i] >= 4) ? 1 : 0;
        }
    }
}

extern "C" void kernel_launch(void* const* d_in, const int* in_sizes, int n_in,
                              void* d_out, int out_size, void* d_ws, size_t ws_size,
                              hipStream_t stream) {
    (void)in_sizes; (void)n_in; (void)out_size; (void)ws_size;
    const float* x    = (const float*)d_in[0];
    const int*   idx1 = (const int*)d_in[1];
    const int*   idx2 = (const int*)d_in[2];
    unsigned short* bm = (unsigned short*)d_ws;
    unsigned short* am = (unsigned short*)((char*)d_ws + 32768);
    int* out = (int*)d_out;

    k_bitmask<<<NIN / 256, 256, 0, stream>>>(x, bm);
    k_layer<1><<<(C * T * BR) / BR_PER_BLK, 256, 0, stream>>>(bm, idx1, am, nullptr);
    k_layer<2><<<(C * BR) / BR_PER_BLK, 256, 0, stream>>>(am, idx2, nullptr, out);
}
